// MultiHeadAttentionBlock_59115929862636
// MI455X (gfx1250) — matrix operand order repaired and checked by reading, not compiled, not run on hardware
//
#include <hip/hip_runtime.h>
#include <hip/hip_bf16.h>

// ---------------------------------------------------------------------------
// MI455X / gfx1250 multi-head attention block (QKV proj + RoPE + flash attn +
// output proj). All matmuls on v_wmma_f32_16x16x32_bf16 (fp32 accumulate).
// Attention K/V staging uses GLOBAL_LOAD_ASYNC_TO_LDS_B128 (ASYNCcnt) with
// double-buffered LDS so the next key block streams in behind the WMMAs.
// ---------------------------------------------------------------------------

typedef __bf16 v16bf __attribute__((ext_vector_type(16)));
typedef float  v8f   __attribute__((ext_vector_type(8)));

constexpr int D_MODEL = 1024;
constexpr int N_SEQ   = 2048;
constexpr int NBATCH  = 2;
constexpr int NHEAD   = 16;
constexpr int DK      = 64;

__device__ __forceinline__ v8f wmma_bf16(v16bf a, v16bf b, v8f c) {
  // (neg_a, A, neg_b, B, c_mod, C, reuse_a, reuse_b)
  return __builtin_amdgcn_wmma_f32_16x16x32_bf16(false, a, false, b,
                                                 (short)0, c, false, false);
}

// A fragment: 16x32 (MxK) bf16 tile, LDS row-major, stride ld.
// lane = M (0..15), half-lane selects K-half {0,16};
// VGPRs 0-7 hold K = kb+0..15 (contiguous, matches B fragments).
__device__ __forceinline__ v16bf frag_a(const __bf16* p, int ld) {
  const int lane = threadIdx.x & 31;
  const int m  = lane & 15;
  const int kb = (lane >> 4) << 4;
  v16bf a;
#pragma unroll
  for (int j = 0; j < 8; ++j) {
    const int k = kb + 2 * j;
    a[2 * j]     = p[m * ld + k];
    a[2 * j + 1] = p[m * ld + k + 1];
  }
  return a;
}

// B fragment from K-major LDS tile: elem(k,n) = p[k*ld + n].
// Lane = N; lanes 0-15 hold K=0..15, lanes 16-31 hold K=16..31 (2 per VGPR).
__device__ __forceinline__ v16bf frag_b_kmaj(const __bf16* p, int ld) {
  const int lane = threadIdx.x & 31;
  const int n  = lane & 15;
  const int kb = (lane >> 4) << 4;
  v16bf b;
#pragma unroll
  for (int j = 0; j < 8; ++j) {
    b[2 * j]     = p[(kb + 2 * j) * ld + n];
    b[2 * j + 1] = p[(kb + 2 * j + 1) * ld + n];
  }
  return b;
}

// B fragment from N-major LDS tile: elem(k,n) = p[n*ld + k] (contiguous in k).
__device__ __forceinline__ v16bf frag_b_nmaj(const __bf16* p, int ld) {
  const int lane = threadIdx.x & 31;
  const int n  = lane & 15;
  const int kb = (lane >> 4) << 4;
  v16bf b;
#pragma unroll
  for (int j = 0; j < 8; ++j) {
    b[2 * j]     = p[n * ld + kb + 2 * j];
    b[2 * j + 1] = p[n * ld + kb + 2 * j + 1];
  }
  return b;
}

// ---------------------------------------------------------------------------
// Async HBM -> LDS copy of one 64x64 bf16 tile (8 KB) with 128 threads.
// Each lane issues 4x GLOBAL_LOAD_ASYNC_TO_LDS_B128 (ASYNCcnt-tracked, no
// VGPR data round-trip). INST_OFFSET is added to BOTH the LDS and global
// addresses per the ISA, so one base pair serves all four beats.
// ---------------------------------------------------------------------------
__device__ __forceinline__ void async_tile_8k(const __bf16* __restrict__ g,
                                              __bf16* l, int tid) {
  const int r = tid >> 1, ch = tid & 1;                 // 64 rows x 2 halves
  const unsigned long long ga =
      (unsigned long long)(uintptr_t)(g + (size_t)r * D_MODEL + ch * 32);
  const unsigned la = (unsigned)(uintptr_t)(l + r * 64 + ch * 32); // LDS offset
  asm volatile("global_load_async_to_lds_b128 %0, %1, off offset:0"
               :: "v"(la), "v"(ga) : "memory");
  asm volatile("global_load_async_to_lds_b128 %0, %1, off offset:16"
               :: "v"(la), "v"(ga) : "memory");
  asm volatile("global_load_async_to_lds_b128 %0, %1, off offset:32"
               :: "v"(la), "v"(ga) : "memory");
  asm volatile("global_load_async_to_lds_b128 %0, %1, off offset:48"
               :: "v"(la), "v"(ga) : "memory");
}

__device__ __forceinline__ void wait_async0() {
  asm volatile("s_wait_asynccnt 0x0" ::: "memory");
}

// ---------------------------------------------------------------------------
// Kernel 1: C[m, n] = sum_k A[m,k] * W[n,k]   (i.e. A @ W^T), optional fused
// RoPE on the epilogue, bf16 output. A, W fp32. 64x64x64 tiles, 4 waves.
// ---------------------------------------------------------------------------
__global__ __launch_bounds__(128)
void qkv_proj_kernel(const float* __restrict__ A, const float* __restrict__ W,
                     __bf16* __restrict__ out, int applyRope) {
  __shared__ __bf16 aLds[64 * 64];   // [m][k]
  __shared__ __bf16 wLds[64 * 64];   // [n][k]  (natural copy of W rows)

  const int tid  = threadIdx.x;
  const int lane = tid & 31;
  const int wave = tid >> 5;
  const int n0 = blockIdx.x * 64;
  const int m0 = blockIdx.y * 64;

  v8f zero = {};
  v8f acc[4] = {zero, zero, zero, zero};

  for (int k0 = 0; k0 < D_MODEL; k0 += 64) {
    __syncthreads();
    {
      const int r = tid >> 1, ch = tid & 1;   // 64 rows x 2 half-rows
      const float4* asrc = (const float4*)(A + (size_t)(m0 + r) * D_MODEL + k0 + ch * 32);
      const float4* wsrc = (const float4*)(W + (size_t)(n0 + r) * D_MODEL + k0 + ch * 32);
      __bf16* ad = aLds + r * 64 + ch * 32;
      __bf16* wd = wLds + r * 64 + ch * 32;
      if (k0 + 64 < D_MODEL) {   // hint next k-panel -> global_prefetch_b8
        __builtin_prefetch(A + (size_t)(m0 + r) * D_MODEL + k0 + 64 + ch * 32, 0, 1);
        __builtin_prefetch(W + (size_t)(n0 + r) * D_MODEL + k0 + 64 + ch * 32, 0, 1);
      }
#pragma unroll
      for (int i = 0; i < 8; ++i) {
        float4 fa = asrc[i];
        ad[4 * i + 0] = (__bf16)fa.x; ad[4 * i + 1] = (__bf16)fa.y;
        ad[4 * i + 2] = (__bf16)fa.z; ad[4 * i + 3] = (__bf16)fa.w;
        float4 fw = wsrc[i];
        wd[4 * i + 0] = (__bf16)fw.x; wd[4 * i + 1] = (__bf16)fw.y;
        wd[4 * i + 2] = (__bf16)fw.z; wd[4 * i + 3] = (__bf16)fw.w;
      }
    }
    __syncthreads();

#pragma unroll
    for (int kk = 0; kk < 64; kk += 32) {
      v16bf af = frag_a(aLds + (wave * 16) * 64 + kk, 64);
#pragma unroll
      for (int nt = 0; nt < 4; ++nt)
        acc[nt] = wmma_bf16(af, frag_b_nmaj(wLds + (nt * 16) * 64 + kk, 64), acc[nt]);
    }
  }

  // Epilogue: C/D layout -> lane holds column n for rows mh..mh+7.
  const int n  = lane & 15;
  const int mh = (lane >> 4) << 3;
#pragma unroll
  for (int nt = 0; nt < 4; ++nt) {
#pragma unroll
    for (int r = 0; r < 8; ++r) {
      const int row = m0 + wave * 16 + mh + r;
      const int col = n0 + nt * 16 + n;
      float val = acc[nt][r];
      if (applyRope) {
        // pair partner (row, col^1) lives in lane^1
        const float vp  = __shfl_xor(val, 1, 32);
        const int   pos = row & (N_SEQ - 1);                 // s within batch
        const float th  = __expf((float)(col >> 1) * (-2.0f / D_MODEL) * 9.210340371976184f);
        const float ang = (float)pos * th;
        const float c = cosf(ang), s = sinf(ang);
        // out[2i] = x0*c + x1*s ; out[2i+1] = x1*c - x0*s  (matches reference)
        val = (col & 1) ? (val * c - vp * s) : (val * c + vp * s);
      }
      out[(size_t)row * D_MODEL + col] = (__bf16)val;
    }
  }
}

// ---------------------------------------------------------------------------
// Kernel 2: flash attention. One (batch, head, 64-query block) per workgroup,
// 4 waves x 16 query rows each. K/V streamed with async-to-LDS into double
// buffers: next key block loads while current block's 16 WMMAs + softmax run.
// ---------------------------------------------------------------------------
__global__ __launch_bounds__(128)
void attn_flash_kernel(const __bf16* __restrict__ Q, const __bf16* __restrict__ K,
                       const __bf16* __restrict__ V, __bf16* __restrict__ O) {
  __shared__ __bf16 qLds[64 * 64];         // [qrow][dk]
  __shared__ __bf16 kLds[2][64 * 64];      // [key][dk], double buffered
  __shared__ __bf16 vLds[2][64 * 64];      // [key][dk], double buffered
  __shared__ __bf16 pLds[4][16 * 64];      // per-wave P scratch [m][key]

  const int tid  = threadIdx.x;
  const int lane = tid & 31;
  const int wave = tid >> 5;
  const int q0    = blockIdx.x * 64;
  const int head  = blockIdx.y;
  const int batch = blockIdx.z;
  const size_t rowBase = (size_t)batch * N_SEQ;
  const int c0 = head * DK;

  // Prologue: async-stage Q block + first K/V block.
  async_tile_8k(Q + (rowBase + q0) * D_MODEL + c0, qLds, tid);
  async_tile_8k(K + rowBase * D_MODEL + c0, kLds[0], tid);
  async_tile_8k(V + rowBase * D_MODEL + c0, vLds[0], tid);
  wait_async0();
  __syncthreads();

  v16bf qf0 = frag_a(qLds + wave * 16 * 64 + 0, 64);
  v16bf qf1 = frag_a(qLds + wave * 16 * 64 + 32, 64);

  v8f zero = {};
  v8f acc[4] = {zero, zero, zero, zero};
  float mrow[8], lrow[8];
#pragma unroll
  for (int r = 0; r < 8; ++r) { mrow[r] = -1e30f; lrow[r] = 0.0f; }

  const int n  = lane & 15;
  const int mh = (lane >> 4) << 3;

  int cur = 0;
  for (int t0 = 0; t0 < N_SEQ; t0 += 64) {
    // Kick off next key block into the shadow buffer (hidden behind compute).
    if (t0 + 64 < N_SEQ) {
      async_tile_8k(K + (rowBase + t0 + 64) * D_MODEL + c0, kLds[cur ^ 1], tid);
      async_tile_8k(V + (rowBase + t0 + 64) * D_MODEL + c0, vLds[cur ^ 1], tid);
    }
    const __bf16* kb = kLds[cur];
    const __bf16* vb = vLds[cur];

    // ---- scores S = (Q @ K^T) / sqrt(dk), 4 key tiles of 16 ----
    v8f sc[4];
    float mblk[8];
#pragma unroll
    for (int r = 0; r < 8; ++r) mblk[r] = -1e30f;

#pragma unroll
    for (int jt = 0; jt < 4; ++jt) {
      v8f s = {};
      s = wmma_bf16(qf0, frag_b_nmaj(kb + (jt * 16) * 64 + 0, 64), s);
      s = wmma_bf16(qf1, frag_b_nmaj(kb + (jt * 16) * 64 + 32, 64), s);
#pragma unroll
      for (int r = 0; r < 8; ++r) {
        float x = s[r] * 0.125f;            // 1/sqrt(64)
        s[r] = x;
        float mx = x;                        // row-max across half-wave lanes
        mx = fmaxf(mx, __shfl_xor(mx, 1, 32));
        mx = fmaxf(mx, __shfl_xor(mx, 2, 32));
        mx = fmaxf(mx, __shfl_xor(mx, 4, 32));
        mx = fmaxf(mx, __shfl_xor(mx, 8, 32));
        mblk[r] = fmaxf(mblk[r], mx);
      }
      sc[jt] = s;
    }

    // ---- online softmax rescale ----
    float mnew[8], alpha[8];
#pragma unroll
    for (int r = 0; r < 8; ++r) {
      mnew[r]  = fmaxf(mrow[r], mblk[r]);
      alpha[r] = __expf(mrow[r] - mnew[r]);
      lrow[r] *= alpha[r];
      mrow[r]  = mnew[r];
    }
#pragma unroll
    for (int nt = 0; nt < 4; ++nt)
#pragma unroll
      for (int r = 0; r < 8; ++r) acc[nt][r] *= alpha[r];

    // ---- P = exp(S - m), row sums, reshape C-layout -> A-layout via LDS ----
#pragma unroll
    for (int jt = 0; jt < 4; ++jt) {
#pragma unroll
      for (int r = 0; r < 8; ++r) {
        float p  = __expf(sc[jt][r] - mnew[r]);
        float ps = p;
        ps += __shfl_xor(ps, 1, 32);
        ps += __shfl_xor(ps, 2, 32);
        ps += __shfl_xor(ps, 4, 32);
        ps += __shfl_xor(ps, 8, 32);
        lrow[r] += ps;
        pLds[wave][(mh + r) * 64 + jt * 16 + n] = (__bf16)p;   // per-wave, in-order
      }
    }

    // ---- O += P @ V ----
#pragma unroll
    for (int kk = 0; kk < 64; kk += 32) {
      v16bf pf = frag_a(pLds[wave] + kk, 64);
#pragma unroll
      for (int nt = 0; nt < 4; ++nt)
        acc[nt] = wmma_bf16(pf, frag_b_kmaj(vb + kk * 64 + nt * 16, 64), acc[nt]);
    }

    // My async stores into the shadow buffer done; barrier also ensures all
    // waves finished reading the current buffer before it becomes the shadow.
    wait_async0();
    __syncthreads();
    cur ^= 1;
  }

  // Epilogue: O / l, heads laid out contiguously in d_model.
#pragma unroll
  for (int nt = 0; nt < 4; ++nt)
#pragma unroll
    for (int r = 0; r < 8; ++r) {
      float val = acc[nt][r] / lrow[r];
      O[(rowBase + q0 + wave * 16 + mh + r) * D_MODEL + c0 + nt * 16 + n] = (__bf16)val;
    }
}

// ---------------------------------------------------------------------------
// Kernel 3: Y = O @ W_o^T + b_o, bf16 x bf16 -> fp32 output.
// ---------------------------------------------------------------------------
__global__ __launch_bounds__(128)
void out_proj_kernel(const __bf16* __restrict__ A, const float* __restrict__ W,
                     const float* __restrict__ bias, float* __restrict__ out) {
  __shared__ __bf16 aLds[64 * 64];
  __shared__ __bf16 wLds[64 * 64];

  const int tid  = threadIdx.x;
  const int lane = tid & 31;
  const int wave = tid >> 5;
  const int n0 = blockIdx.x * 64;
  const int m0 = blockIdx.y * 64;

  v8f zero = {};
  v8f acc[4] = {zero, zero, zero, zero};

  for (int k0 = 0; k0 < D_MODEL; k0 += 64) {
    __syncthreads();
    {
      const int r = tid >> 1, ch = tid & 1;
      const uint4* asrc = (const uint4*)(A + (size_t)(m0 + r) * D_MODEL + k0 + ch * 32);
      uint4* ad = (uint4*)(aLds + r * 64 + ch * 32);
#pragma unroll
      for (int i = 0; i < 4; ++i) ad[i] = asrc[i];
      const float4* wsrc = (const float4*)(W + (size_t)(n0 + r) * D_MODEL + k0 + ch * 32);
      __bf16* wd = wLds + r * 64 + ch * 32;
#pragma unroll
      for (int i = 0; i < 8; ++i) {
        float4 f = wsrc[i];
        wd[4 * i + 0] = (__bf16)f.x; wd[4 * i + 1] = (__bf16)f.y;
        wd[4 * i + 2] = (__bf16)f.z; wd[4 * i + 3] = (__bf16)f.w;
      }
    }
    __syncthreads();
#pragma unroll
    for (int kk = 0; kk < 64; kk += 32) {
      v16bf af = frag_a(aLds + (wave * 16) * 64 + kk, 64);
#pragma unroll
      for (int nt = 0; nt < 4; ++nt)
        acc[nt] = wmma_bf16(af, frag_b_nmaj(wLds + (nt * 16) * 64 + kk, 64), acc[nt]);
    }
  }

  const int n  = lane & 15;
  const int mh = (lane >> 4) << 3;
#pragma unroll
  for (int nt = 0; nt < 4; ++nt)
#pragma unroll
    for (int r = 0; r < 8; ++r) {
      const int row = m0 + wave * 16 + mh + r;
      const int col = n0 + nt * 16 + n;
      out[(size_t)row * D_MODEL + col] = acc[nt][r] + bias[col];
    }
}

// ---------------------------------------------------------------------------
extern "C" void kernel_launch(void* const* d_in, const int* in_sizes, int n_in,
                              void* d_out, int out_size, void* d_ws, size_t ws_size,
                              hipStream_t stream) {
  (void)in_sizes; (void)n_in; (void)out_size; (void)ws_size;

  const float* q   = (const float*)d_in[0];
  const float* k   = (const float*)d_in[1];
  const float* v   = (const float*)d_in[2];
  const float* W_q = (const float*)d_in[3];
  const float* W_k = (const float*)d_in[4];
  const float* W_v = (const float*)d_in[5];
  const float* W_o = (const float*)d_in[6];
  const float* b_o = (const float*)d_in[7];
  float* out = (float*)d_out;

  const size_t elems = (size_t)NBATCH * N_SEQ * D_MODEL;   // 4 Mi elements
  __bf16* Qb = (__bf16*)d_ws;          // 8 MB each, 32 MB total workspace
  __bf16* Kb = Qb + elems;
  __bf16* Vb = Kb + elems;
  __bf16* Ob = Vb + elems;

  const dim3 gProj(D_MODEL / 64, (NBATCH * N_SEQ) / 64);   // (16, 64)
  const dim3 blk(128);

  qkv_proj_kernel<<<gProj, blk, 0, stream>>>(q, W_q, Qb, 1);
  qkv_proj_kernel<<<gProj, blk, 0, stream>>>(k, W_k, Kb, 1);
  qkv_proj_kernel<<<gProj, blk, 0, stream>>>(v, W_v, Vb, 0);

  const dim3 gAttn(N_SEQ / 64, NHEAD, NBATCH);             // (32, 16, 2)
  attn_flash_kernel<<<gAttn, blk, 0, stream>>>(Qb, Kb, Vb, Ob);

  out_proj_kernel<<<gProj, blk, 0, stream>>>(Ob, W_o, b_o, out);
}
